// CrossAttentionBlock_54666343743582
// MI455X (gfx1250) — matrix operand order repaired and checked by reading, not compiled, not run on hardware
//
#include <hip/hip_runtime.h>
#include <cstdint>

// ---------------- problem constants ----------------
#define BB     4
#define HH     32
#define WW     32
#define MM     16
#define NN     16384           // H*W*M
#define DD     128
#define HEADS  4
#define HK     32              // D / HEADS
#define D2     256
#define D4     512
#define EPSF   1e-5f

// ---------------- bf16 helpers ----------------
typedef __attribute__((ext_vector_type(16))) __bf16 v16bf;
typedef __attribute__((ext_vector_type(8)))  float  v8f;

__device__ __forceinline__ unsigned short f2bfbits(float f) {
  union { float f; uint32_t u; } c; c.f = f;
  uint32_t u = c.u;
  uint32_t r = (u + 0x7fffu + ((u >> 16) & 1u)) >> 16;
  return (unsigned short)r;
}
__device__ __forceinline__ float bf2f(unsigned short s) {
  union { uint32_t u; float f; } c; c.u = ((uint32_t)s) << 16;
  return c.f;
}
__device__ __forceinline__ __bf16 bits2bf(unsigned short s) {
  return __builtin_bit_cast(__bf16, s);
}

// A fragment: 16x32 bf16 tile, row-major in LDS (ldk halfwords per row).
// Lane layout: lanes 0-15 row M=lane hold K = 0..15;
// lanes 16-31 row M=lane-16 hold K = 16..31.
__device__ __forceinline__ v16bf load_a_frag(const unsigned short* At, int ldk,
                                             int kbase, int lane) {
  int m = lane & 15;
  int kh = (lane & 16) ? 16 : 0;
  const unsigned short* row = At + m * ldk + kbase;
  v16bf a;
#pragma unroll
  for (int e = 0; e < 8; ++e) a[e] = bits2bf(row[kh + e]);
#pragma unroll
  for (int e = 0; e < 8; ++e) a[8 + e] = bits2bf(row[kh + 8 + e]);
  return a;
}

// B fragment: 32x16 (K x N). B[k][n] = W[(kbase+k)*ldn + nbase + n] (fp32 weights).
// Lanes 0-15: column n=lane, K 0..15 ; lanes 16-31: column n=lane-16, K 16..31.
__device__ __forceinline__ v16bf load_b_frag(const float* W, int ldn,
                                             int kbase, int nbase, int lane) {
  int n = lane & 15;
  int ko = (lane & 16) ? 16 : 0;
  v16bf b;
#pragma unroll
  for (int e = 0; e < 16; ++e)
    b[e] = bits2bf(f2bfbits(W[(size_t)(kbase + ko + e) * ldn + nbase + n]));
  return b;
}

// Transposed-B fragment: B[k][n] = W[(nbase+n)*ldk + kbase + k]   (rep_w case)
__device__ __forceinline__ v16bf load_b_fragT(const float* W, int ldk,
                                              int kbase, int nbase, int lane) {
  int n = lane & 15;
  int ko = (lane & 16) ? 16 : 0;
  const float* col = W + (size_t)(nbase + n) * ldk + kbase + ko;
  v16bf b;
#pragma unroll
  for (int e = 0; e < 16; ++e) b[e] = bits2bf(f2bfbits(col[e]));
  return b;
}

// ---------------- K1: LayerNorm(x1), LayerNorm(x2) -> bf16 ----------------
__global__ __launch_bounds__(256) void ln1_kernel(
    const float* __restrict__ x1, const float* __restrict__ x2,
    const float* __restrict__ g, const float* __restrict__ bta,
    unsigned short* __restrict__ n1, unsigned short* __restrict__ n2) {
  int gwave = (blockIdx.x * 256 + threadIdx.x) >> 5;   // 0 .. 2*B*N-1
  int lane  = threadIdx.x & 31;
  int strm  = gwave >> 16;                              // B*N = 65536 rows each
  int row   = gwave & 0xFFFF;
  const float* x = strm ? x2 : x1;
  unsigned short* o = strm ? n2 : n1;
  float4 v = *reinterpret_cast<const float4*>(x + (size_t)row * DD + lane * 4);
  float s  = v.x + v.y + v.z + v.w;
  float sq = v.x * v.x + v.y * v.y + v.z * v.z + v.w * v.w;
#pragma unroll
  for (int m = 16; m >= 1; m >>= 1) {
    s  += __shfl_xor(s,  m, 32);
    sq += __shfl_xor(sq, m, 32);
  }
  float mu   = s * (1.f / 128.f);
  float var  = sq * (1.f / 128.f) - mu * mu;
  float rstd = rsqrtf(var + EPSF);
  unsigned short* op = o + (size_t)row * DD + lane * 4;
  const float* gp = g + lane * 4;
  const float* bp = bta + lane * 4;
  float vv[4] = {v.x, v.y, v.z, v.w};
#pragma unroll
  for (int j = 0; j < 4; ++j)
    op[j] = f2bfbits((vv[j] - mu) * rstd * gp[j] + bp[j]);
}

// ---------------- K2: per (b, channel) softmax stats over N tokens --------
__global__ __launch_bounds__(256) void keystats_kernel(
    const unsigned short* __restrict__ n2,
    float* __restrict__ kmax, float* __restrict__ kz) {
  int bd = blockIdx.x;            // 0..511
  int b = bd >> 7, d = bd & 127;
  const unsigned short* base = n2 + (size_t)b * NN * DD + d;
  __shared__ float red[256];
  int t = threadIdx.x;
  float mx = -1e30f;
  for (int n = t; n < NN; n += 256) mx = fmaxf(mx, bf2f(base[(size_t)n * DD]));
  red[t] = mx; __syncthreads();
  for (int s = 128; s > 0; s >>= 1) {
    if (t < s) red[t] = fmaxf(red[t], red[t + s]);
    __syncthreads();
  }
  mx = red[0]; __syncthreads();
  float se = 0.f;
  for (int n = t; n < NN; n += 256) se += __expf(bf2f(base[(size_t)n * DD]) - mx);
  red[t] = se; __syncthreads();
  for (int s = 128; s > 0; s >>= 1) {
    if (t < s) red[t] += red[t + s];
    __syncthreads();
  }
  if (t == 0) { kmax[bd] = mx; kz[bd] = red[0]; }
}

// ---------------- K3: context[b,h,k,v] = sum_n key[k,n] * v1[v,n] ---------
__global__ __launch_bounds__(256) void context_kernel(
    const unsigned short* __restrict__ n1, const unsigned short* __restrict__ n2,
    const float* __restrict__ kmax, const float* __restrict__ kz,
    float* __restrict__ ctx) {
  const int CHUNKS = 64;               // 256 tokens per chunk
  int bh = blockIdx.x / CHUNKS;
  int chunk = blockIdx.x % CHUNKS;
  int b = bh >> 2, h = bh & 3;
  int n0 = chunk * 256;
  __shared__ float sk[32][33];
  __shared__ float sv[32][33];
  int t = threadIdx.x;
  int myk = t >> 3;
  int vb  = (t & 7) * 4;
  int c    = t & 31;                   // loader channel
  int jrow = t >> 5;                   // loader token sub-row 0..7
  float km  = kmax[b * 128 + h * 32 + c];
  float kzi = 1.f / kz[b * 128 + h * 32 + c];
  float acc0 = 0, acc1 = 0, acc2 = 0, acc3 = 0;
  for (int nn = 0; nn < 256; nn += 32) {
#pragma unroll
    for (int jj = 0; jj < 32; jj += 8) {
      int tok = n0 + nn + jrow + jj;
      size_t idx = ((size_t)(b * NN + tok)) * DD + h * 32 + c;
      sk[c][jrow + jj] = __expf(bf2f(n2[idx]) - km) * kzi;
      sv[c][jrow + jj] = bf2f(n1[idx]);
    }
    __syncthreads();
#pragma unroll
    for (int j = 0; j < 32; ++j) {
      float kf = sk[myk][j];
      acc0 += kf * sv[vb + 0][j];
      acc1 += kf * sv[vb + 1][j];
      acc2 += kf * sv[vb + 2][j];
      acc3 += kf * sv[vb + 3][j];
    }
    __syncthreads();
  }
  float* cb = ctx + ((size_t)(bh * 32 + myk)) * 32 + vb;
  atomicAdd(cb + 0, acc0);
  atomicAdd(cb + 1, acc1);
  atomicAdd(cb + 2, acc2);
  atomicAdd(cb + 3, acc3);
}

// ---------------- K4: att -> agg[b,n,d] bf16 (token-major) ----------------
__global__ __launch_bounds__(256) void att_kernel(
    const unsigned short* __restrict__ n2, const float* __restrict__ ctx,
    unsigned short* __restrict__ aggT) {
  int tkn = blockIdx.x * 256 + threadIdx.x;     // 0..65535; whole block same b
  int b = tkn >> 14;
  __shared__ float sctx[HEADS * HK * HK];       // 16KB
  for (int i = threadIdx.x; i < HEADS * HK * HK; i += 256)
    sctx[i] = ctx[(size_t)b * HEADS * HK * HK + i];
  __syncthreads();
  const unsigned short* row = n2 + (size_t)tkn * DD;
  unsigned short* orow = aggT + (size_t)tkn * DD;
#pragma unroll
  for (int h = 0; h < HEADS; ++h) {
    float q[HK];
    float mx = -1e30f;
#pragma unroll
    for (int k = 0; k < HK; ++k) { q[k] = bf2f(row[h * HK + k]); mx = fmaxf(mx, q[k]); }
    float s = 0.f;
#pragma unroll
    for (int k = 0; k < HK; ++k) { q[k] = __expf(q[k] - mx); s += q[k]; }
    float inv = 1.f / s;
    float acc[HK];
#pragma unroll
    for (int v = 0; v < HK; ++v) acc[v] = 0.f;
    const float* ch = sctx + h * HK * HK;
#pragma unroll 4
    for (int k = 0; k < HK; ++k) {
      float qk = q[k] * inv;
#pragma unroll
      for (int v = 0; v < HK; ++v) acc[v] += ch[k * HK + v] * qk;
    }
#pragma unroll
    for (int v = 0; v < HK; ++v) orow[h * HK + v] = f2bfbits(acc[v]);
  }
}

// ------ K5: rep GEMM (WMMA) + bias + aln LN + residual + ln2 -> tx, y -----
__global__ __launch_bounds__(256) void repln_kernel(
    const unsigned short* __restrict__ aggT, const float* __restrict__ rep_w,
    const float* __restrict__ rep_b, const float* __restrict__ aln_g,
    const float* __restrict__ aln_b, const float* __restrict__ x1,
    const float* __restrict__ x2, const float* __restrict__ ln2_g,
    const float* __restrict__ ln2_b, float* __restrict__ tx,
    unsigned short* __restrict__ y) {
  size_t tok0 = (size_t)blockIdx.x * 16;
  int t = threadIdx.x, lane = t & 31, wv = t >> 5;
  __shared__ unsigned short sA[16 * DD];   // 4KB agg tile
  __shared__ float sC[16 * D2];            // 16KB output tile
  __shared__ float sMu[16], sRs[16];
  for (int i = t; i < 16 * DD; i += 256) sA[i] = aggT[tok0 * DD + i];
  __syncthreads();
#pragma unroll
  for (int tt = 0; tt < 2; ++tt) {
    int nb = wv * 32 + tt * 16;
    v8f acc = {};
#pragma unroll
    for (int kk = 0; kk < 4; ++kk) {
      v16bf af = load_a_frag(sA, DD, kk * 32, lane);
      v16bf bf = load_b_fragT(rep_w, DD, kk * 32, nb, lane);
      acc = __builtin_amdgcn_wmma_f32_16x16x32_bf16(false, af, false, bf,
                                                    (short)0, acc, false, false);
    }
    int mrow = (lane & 16) ? 8 : 0;
    int ncol = nb + (lane & 15);
    float bias = rep_b[ncol];
#pragma unroll
    for (int r = 0; r < 8; ++r) sC[(mrow + r) * D2 + ncol] = acc[r] + bias;
  }
  __syncthreads();
  // aln LayerNorm stats (16 lanes per token)
  {
    int tok = t >> 4, l16 = t & 15;
    float s = 0, sq = 0;
    for (int c = l16; c < D2; c += 16) { float v = sC[tok * D2 + c]; s += v; sq += v * v; }
#pragma unroll
    for (int m = 8; m >= 1; m >>= 1) { s += __shfl_xor(s, m, 32); sq += __shfl_xor(sq, m, 32); }
    if (l16 == 0) {
      float mu = s * (1.f / 256.f);
      float var = sq * (1.f / 256.f) - mu * mu;
      sMu[tok] = mu; sRs[tok] = rsqrtf(var + EPSF);
    }
  }
  __syncthreads();
  // tx = concat(x1,x2) + aln(rep);  stash tx back into sC for ln2
  for (int i = t; i < 16 * D2; i += 256) {
    int tok = i >> 8, c = i & 255;
    size_t gt = tok0 + tok;
    float a = (sC[i] - sMu[tok]) * sRs[tok] * aln_g[c] + aln_b[c];
    float xv = (c < DD) ? x1[gt * DD + c] : x2[gt * DD + (c - DD)];
    float txv = a + xv;
    tx[gt * D2 + c] = txv;
    sC[i] = txv;
  }
  __syncthreads();
  {
    int tok = t >> 4, l16 = t & 15;
    float s = 0, sq = 0;
    for (int c = l16; c < D2; c += 16) { float v = sC[tok * D2 + c]; s += v; sq += v * v; }
#pragma unroll
    for (int m = 8; m >= 1; m >>= 1) { s += __shfl_xor(s, m, 32); sq += __shfl_xor(sq, m, 32); }
    if (l16 == 0) {
      float mu = s * (1.f / 256.f);
      float var = sq * (1.f / 256.f) - mu * mu;
      sMu[tok] = mu; sRs[tok] = rsqrtf(var + EPSF);
    }
  }
  __syncthreads();
  for (int i = t; i < 16 * D2; i += 256) {
    int tok = i >> 8, c = i & 255;
    float v = (sC[i] - sMu[tok]) * sRs[tok] * ln2_g[c] + ln2_b[c];
    y[(tok0 + tok) * D2 + c] = f2bfbits(v);
  }
}

// ---------------- K6: fc1 GEMM (WMMA) -> h1 bf16 --------------------------
__global__ __launch_bounds__(256) void fc1_kernel(
    const unsigned short* __restrict__ y, const float* __restrict__ fc1_w,
    const float* __restrict__ fc1_b, unsigned short* __restrict__ h1) {
  size_t tok0 = (size_t)blockIdx.x * 16;
  int t = threadIdx.x, lane = t & 31, wv = t >> 5;
  __shared__ unsigned short sA[16 * D2];   // 8KB
  for (int i = t; i < 16 * D2; i += 256) sA[i] = y[tok0 * D2 + i];
  __syncthreads();
#pragma unroll
  for (int tt = 0; tt < 4; ++tt) {
    int nb = wv * 64 + tt * 16;
    v8f acc = {};
#pragma unroll
    for (int kk = 0; kk < 8; ++kk) {
      v16bf af = load_a_frag(sA, D2, kk * 32, lane);
      v16bf bf = load_b_frag(fc1_w, D4, kk * 32, nb, lane);
      acc = __builtin_amdgcn_wmma_f32_16x16x32_bf16(false, af, false, bf,
                                                    (short)0, acc, false, false);
    }
    int mrow = (lane & 16) ? 8 : 0;
    int ncol = nb + (lane & 15);
    float bias = fc1_b[ncol];
#pragma unroll
    for (int r = 0; r < 8; ++r)
      h1[(tok0 + mrow + r) * D4 + ncol] = f2bfbits(acc[r] + bias);
  }
}

// ------- K7: depthwise 3x3x3 conv + residual + mln LN + GELU -> ax --------
__global__ __launch_bounds__(256) void dw_kernel(
    const unsigned short* __restrict__ h1, const float* __restrict__ dw_w,
    const float* __restrict__ dw_b, const float* __restrict__ mln_g,
    const float* __restrict__ mln_b, unsigned short* __restrict__ ax) {
  int tkn = blockIdx.x;                 // 0..65535
  int b = tkn >> 14;
  int n = tkn & 16383;
  int hh = n >> 9, wc = (n >> 4) & 31, mm = n & 15;
  int t = threadIdx.x;
  int c0 = t * 2;
  const unsigned short* hb = h1 + (size_t)b * NN * D4;
  float v0 = 0.f, v1 = 0.f, h1c0 = 0.f, h1c1 = 0.f;
  const float* w0 = dw_w + (size_t)c0 * 27;
  const float* w1 = dw_w + (size_t)(c0 + 1) * 27;
#pragma unroll
  for (int kd = 0; kd < 3; ++kd) {
    int hz = hh + kd - 1;
    if ((unsigned)hz >= HH) continue;
#pragma unroll
    for (int kh = 0; kh < 3; ++kh) {
      int wz = wc + kh - 1;
      if ((unsigned)wz >= WW) continue;
#pragma unroll
      for (int kw = 0; kw < 3; ++kw) {
        int mz = mm + kw - 1;
        if ((unsigned)mz >= MM) continue;
        size_t nn = (size_t)hz * 512 + (size_t)wz * 16 + mz;
        uint32_t pk = *reinterpret_cast<const uint32_t*>(hb + nn * D4 + c0);
        float a0 = bf2f((unsigned short)(pk & 0xFFFF));
        float a1 = bf2f((unsigned short)(pk >> 16));
        int widx = kd * 9 + kh * 3 + kw;
        v0 += a0 * w0[widx];
        v1 += a1 * w1[widx];
        if (kd == 1 && kh == 1 && kw == 1) { h1c0 = a0; h1c1 = a1; }
      }
    }
  }
  float val0 = v0 + dw_b[c0] + h1c0;
  float val1 = v1 + dw_b[c0 + 1] + h1c1;
  __shared__ float r1[256], r2[256];
  r1[t] = val0 + val1;
  r2[t] = val0 * val0 + val1 * val1;
  __syncthreads();
  for (int s = 128; s > 0; s >>= 1) {
    if (t < s) { r1[t] += r1[t + s]; r2[t] += r2[t + s]; }
    __syncthreads();
  }
  float mu = r1[0] * (1.f / 512.f);
  float var = r2[0] * (1.f / 512.f) - mu * mu;
  float rstd = rsqrtf(var + EPSF);
  float g0 = (val0 - mu) * rstd * mln_g[c0] + mln_b[c0];
  float g1 = (val1 - mu) * rstd * mln_g[c0 + 1] + mln_b[c0 + 1];
  g0 = 0.5f * g0 * (1.f + erff(g0 * 0.70710678118654752f));
  g1 = 0.5f * g1 * (1.f + erff(g1 * 0.70710678118654752f));
  unsigned short* op = ax + (size_t)tkn * D4 + c0;
  op[0] = f2bfbits(g0);
  op[1] = f2bfbits(g1);
}

// ---------------- K8: fc2 GEMM (WMMA) + tx residual -> out ----------------
__global__ __launch_bounds__(256) void fc2_kernel(
    const unsigned short* __restrict__ ax, const float* __restrict__ fc2_w,
    const float* __restrict__ fc2_b, const float* __restrict__ tx,
    float* __restrict__ out) {
  size_t tok0 = (size_t)blockIdx.x * 16;
  int t = threadIdx.x, lane = t & 31, wv = t >> 5;
  __shared__ unsigned short sA[16 * D4];   // 16KB
  for (int i = t; i < 16 * D4; i += 256) sA[i] = ax[tok0 * D4 + i];
  __syncthreads();
#pragma unroll
  for (int tt = 0; tt < 2; ++tt) {
    int nb = wv * 32 + tt * 16;
    v8f acc = {};
#pragma unroll
    for (int kk = 0; kk < 16; ++kk) {
      v16bf af = load_a_frag(sA, D4, kk * 32, lane);
      v16bf bf = load_b_frag(fc2_w, D2, kk * 32, nb, lane);
      acc = __builtin_amdgcn_wmma_f32_16x16x32_bf16(false, af, false, bf,
                                                    (short)0, acc, false, false);
    }
    int mrow = (lane & 16) ? 8 : 0;
    int ncol = nb + (lane & 15);
    float bias = fc2_b[ncol];
#pragma unroll
    for (int r = 0; r < 8; ++r) {
      size_t gt = tok0 + mrow + r;
      out[gt * D2 + ncol] = tx[gt * D2 + ncol] + acc[r] + bias;
    }
  }
}

// ---------------- host: launch sequence -----------------------------------
extern "C" void kernel_launch(void* const* d_in, const int* in_sizes, int n_in,
                              void* d_out, int out_size, void* d_ws, size_t ws_size,
                              hipStream_t stream) {
  const float* x1    = (const float*)d_in[0];
  const float* x2    = (const float*)d_in[1];
  const float* ln1_g = (const float*)d_in[2];
  const float* ln1_b = (const float*)d_in[3];
  const float* rep_w = (const float*)d_in[4];
  const float* rep_b = (const float*)d_in[5];
  const float* aln_g = (const float*)d_in[6];
  const float* aln_b = (const float*)d_in[7];
  const float* ln2_g = (const float*)d_in[8];
  const float* ln2_b = (const float*)d_in[9];
  const float* fc1_w = (const float*)d_in[10];
  const float* fc1_b = (const float*)d_in[11];
  const float* dw_w  = (const float*)d_in[12];
  const float* dw_b  = (const float*)d_in[13];
  const float* mln_g = (const float*)d_in[14];
  const float* mln_b = (const float*)d_in[15];
  const float* fc2_w = (const float*)d_in[16];
  const float* fc2_b = (const float*)d_in[17];
  float* out = (float*)d_out;

  uint8_t* ws = (uint8_t*)d_ws;
  // segment A (67.1MB): n1 | n2 | agg early; y aliases n1+n2; ax aliases whole A
  unsigned short* n1  = (unsigned short*)(ws + 0);           // 16.78MB
  unsigned short* n2  = (unsigned short*)(ws + 16777216);    // 16.78MB
  unsigned short* agg = (unsigned short*)(ws + 33554432);    // 16.78MB
  unsigned short* y   = (unsigned short*)(ws + 0);           // 33.55MB (n1,n2 dead)
  unsigned short* ax  = (unsigned short*)(ws + 0);           // 67.1MB  (A dead)
  float*          tx  = (float*)(ws + 67108864);             // 67.1MB
  unsigned short* h1  = (unsigned short*)(ws + 134217728);   // 67.1MB
  float*          ctx = (float*)(ws + 201326592);            // 64KB
  float*          kmx = (float*)(ws + 201392128);            // 2KB
  float*          kzv = (float*)(ws + 201394176);            // 2KB

  // 1) LN of both streams: 131072 rows, 8 rows/block
  ln1_kernel<<<16384, 256, 0, stream>>>(x1, x2, ln1_g, ln1_b, n1, n2);
  // 2) key softmax stats over tokens
  keystats_kernel<<<BB * DD, 256, 0, stream>>>(n2, kmx, kzv);
  // 3) context = key @ V^T  (zero accumulators first; ws is not re-poisoned)
  hipMemsetAsync(ctx, 0, (size_t)BB * HEADS * HK * HK * sizeof(float), stream);
  context_kernel<<<BB * HEADS * 64, 256, 0, stream>>>(n1, n2, kmx, kzv, ctx);
  // 4) att = context^T @ softmax_k(Q)  -> agg (token-major)
  att_kernel<<<BB * NN / 256, 256, 0, stream>>>(n2, ctx, agg);
  // 5) rep GEMM + aln LN + residual + ln2  (WMMA)
  repln_kernel<<<BB * NN / 16, 256, 0, stream>>>(agg, rep_w, rep_b, aln_g, aln_b,
                                                 x1, x2, ln2_g, ln2_b, tx, y);
  // 6) fc1 GEMM (WMMA)
  fc1_kernel<<<BB * NN / 16, 256, 0, stream>>>(y, fc1_w, fc1_b, h1);
  // 7) depthwise conv + mln LN + GELU
  dw_kernel<<<BB * NN, 256, 0, stream>>>(h1, dw_w, dw_b, mln_g, mln_b, ax);
  // 8) fc2 GEMM + residual (WMMA)
  fc2_kernel<<<BB * NN / 16, 256, 0, stream>>>(ax, fc2_w, fc2_b, tx, out);
}
